// Inference_Multilabel_Loss_13357348290933
// MI455X (gfx1250) — compile-verified, matrix operand order checked by reading, not run on hardware
//
#include <hip/hip_runtime.h>
#include <hip/hip_bf16.h>
#include <cstdint>

#define TEMPERATURE 0.07f
#define ALPHA       0.6f
#define K_PENALTY   0.01f
#define NUM_ITERS   8
#define NCLS        1000
#define NPAIR       2000
#define NROWS       8192

typedef __attribute__((ext_vector_type(16))) _Float16 v16h;
typedef __attribute__((ext_vector_type(8)))  float    v8f;

// ---------------------------------------------------------------------------
// Kernel 1: sim = (features @ text_features^T) / TEMPERATURE via WMMA.
// The reference computes this then deletes it; it is dead w.r.t. the output,
// so we compute it in f16 (precision irrelevant) into workspace. One wave per
// 16x16 output tile, K=512 marched in 32-wide WMMA steps.
// A (16x32 f16) lane layout: lanes 0-15 hold K={0..7,16..23}, lanes 16-31
// hold K={8..15,24..31}; row = lane&15.  B mirrors with col = lane&15.
// ---------------------------------------------------------------------------
__global__ __launch_bounds__(128) void sim_wmma_kernel(
    const float* __restrict__ F,   // 256 x 512
    const float* __restrict__ T,   // 1000 x 512
    float* __restrict__ S)         // 256 x 1000 (workspace)
{
  const int lane = threadIdx.x & 31;
  const int wave = threadIdx.x >> 5;
  const int tile = blockIdx.x * 4 + wave;       // 16 m-tiles x 63 n-tiles
  if (tile >= 16 * 63) return;                  // wave-uniform: EXEC stays full
  const int mt = tile & 15;
  const int nt = tile >> 4;
  const int m  = mt * 16 + (lane & 15);
  const int n  = nt * 16 + (lane & 15);
  const int kb = (lane < 16) ? 0 : 8;
  const bool nv = (n < NCLS);
  const float* arow = F + (size_t)m * 512 + kb;
  const float* brow = T + (size_t)(nv ? n : 0) * 512 + kb;

  v8f acc = {};
  for (int kk = 0; kk < 512; kk += 32) {
    const float4 a0 = *(const float4*)(arow + kk + 0);
    const float4 a1 = *(const float4*)(arow + kk + 4);
    const float4 a2 = *(const float4*)(arow + kk + 16);
    const float4 a3 = *(const float4*)(arow + kk + 20);
    const float4 b0 = *(const float4*)(brow + kk + 0);
    const float4 b1 = *(const float4*)(brow + kk + 4);
    const float4 b2 = *(const float4*)(brow + kk + 16);
    const float4 b3 = *(const float4*)(brow + kk + 20);
    v16h af, bf;
    af[0]=(_Float16)a0.x;  af[1]=(_Float16)a0.y;  af[2]=(_Float16)a0.z;  af[3]=(_Float16)a0.w;
    af[4]=(_Float16)a1.x;  af[5]=(_Float16)a1.y;  af[6]=(_Float16)a1.z;  af[7]=(_Float16)a1.w;
    af[8]=(_Float16)a2.x;  af[9]=(_Float16)a2.y;  af[10]=(_Float16)a2.z; af[11]=(_Float16)a2.w;
    af[12]=(_Float16)a3.x; af[13]=(_Float16)a3.y; af[14]=(_Float16)a3.z; af[15]=(_Float16)a3.w;
    bf[0]=(_Float16)b0.x;  bf[1]=(_Float16)b0.y;  bf[2]=(_Float16)b0.z;  bf[3]=(_Float16)b0.w;
    bf[4]=(_Float16)b1.x;  bf[5]=(_Float16)b1.y;  bf[6]=(_Float16)b1.z;  bf[7]=(_Float16)b1.w;
    bf[8]=(_Float16)b2.x;  bf[9]=(_Float16)b2.y;  bf[10]=(_Float16)b2.z; bf[11]=(_Float16)b2.w;
    bf[12]=(_Float16)b3.x; bf[13]=(_Float16)b3.y; bf[14]=(_Float16)b3.z; bf[15]=(_Float16)b3.w;
    acc = __builtin_amdgcn_wmma_f32_16x16x32_f16(
        /*neg_a=*/false, af, /*neg_b=*/false, bf,
        /*c_mod=*/(short)0, acc, /*reuse_a=*/false, /*reuse_b=*/false);
  }

  if (nv) {
    const float sc = 1.0f / TEMPERATURE;
    const int rb = mt * 16 + ((lane < 16) ? 0 : 8);  // C layout: VGPR r -> M = r (+8 hi lanes)
#pragma unroll
    for (int r = 0; r < 8; ++r)
      S[(size_t)(rb + r) * NCLS + n] = acc[r] * sc;
  }
}

// ---------------------------------------------------------------------------
// Kernel 2: the real computation. One 256-thread block per batch row.
// State: a[4]/b[4] in registers (thread owns columns tid, tid+256, ...).
// LDS: q0 (gather target), integer penalty counters (deterministic scatter),
// and the pair table staged via CDNA5 async global->LDS copies.
// ---------------------------------------------------------------------------
__global__ __launch_bounds__(256) void sinkhorn_kernel(
    const float* __restrict__ simw,   // 8192 x 1000 x 2
    const int*   __restrict__ pairs,  // 2000 x 2
    float*       __restrict__ out)    // 8192 x 1000 x 2
{
  __shared__ __align__(16) int spairs[2 * NPAIR];   // 16000 B
  __shared__ float    q0s[NCLS];                    //  4000 B
  __shared__ unsigned cnts[NCLS];                   //  4000 B

  const int tid = threadIdx.x;
  const int row = blockIdx.x;
  const float* src = simw + (size_t)row * (2 * NCLS);
  float*       dst = out  + (size_t)row * (2 * NCLS);

  // Stage pair table (1000 x 16B chunks) into LDS asynchronously, overlapped
  // with the exp() of this row's Q0.
  {
    const unsigned lds_base = (unsigned)(uintptr_t)(&spairs[0]);
#pragma unroll
    for (int r = 0; r < 4; ++r) {
      const int chunk = tid + r * 256;
      if (chunk < (2 * NPAIR * 4) / 16) {
        const unsigned lds_addr = lds_base + (unsigned)chunk * 16u;
        const unsigned goff     = (unsigned)chunk * 16u;
        asm volatile("global_load_async_to_lds_b128 %0, %1, %2"
                     :: "v"(lds_addr), "v"(goff), "s"(pairs) : "memory");
      }
    }
  }

  // Q0 = exp(sim_matrix_whole[row]) ; coalesced float2 per column.
  float a[4], b[4];
#pragma unroll
  for (int j = 0; j < 4; ++j) {
    const int c = tid + j * 256;
    if (c < NCLS) {
      const float2 v = ((const float2*)src)[c];
      a[j] = __expf(v.x);
      b[j] = __expf(v.y);
    } else { a[j] = 0.0f; b[j] = 0.0f; }
  }

  asm volatile("s_wait_asynccnt 0" ::: "memory");
  __syncthreads();

  // Pull this thread's ~8 pairs from LDS into registers (reused every iter).
  int p1[8], p2[8];
#pragma unroll
  for (int k = 0; k < 8; ++k) {
    const int p = tid + k * 256;
    if (p < NPAIR) { p1[k] = spairs[2 * p]; p2[k] = spairs[2 * p + 1]; }
    else           { p1[k] = -1;            p2[k] = -1; }
  }

  for (int it = 0; it < NUM_ITERS; ++it) {
    // Normalize; publish q0; clear counters.
#pragma unroll
    for (int j = 0; j < 4; ++j) {
      const int c = tid + j * 256;
      if (c < NCLS) {
        const float inv = 1.0f / (a[j] + b[j]);
        a[j] *= inv;                 // q0
        b[j] *= inv;                 // q1
        q0s[c]  = a[j];
        cnts[c] = 0u;
      }
    }
    __syncthreads();

    // Pair test + deterministic integer scatter (ds_add_u32).
#pragma unroll
    for (int k = 0; k < 8; ++k) {
      if (p1[k] >= 0) {
        const float ps = q0s[p1[k]] + q0s[p2[k]];
        if (ps < ALPHA) {
          atomicAdd(&cnts[p1[k]], 1u);
          atomicAdd(&cnts[p2[k]], 1u);
        }
      }
    }
    __syncthreads();

    // Q[:, :, 0] *= exp(M),  M = K_PENALTY * count  (exact, order-free).
#pragma unroll
    for (int j = 0; j < 4; ++j) {
      const int c = tid + j * 256;
      if (c < NCLS) a[j] *= __expf(K_PENALTY * (float)cnts[c]);
    }
    __syncthreads();   // protect cnts/q0s reuse next iteration
  }

  // Final Q = (q0 * exp(M), q1) — coalesced float2 store.
#pragma unroll
  for (int j = 0; j < 4; ++j) {
    const int c = tid + j * 256;
    if (c < NCLS) {
      float2 v; v.x = a[j]; v.y = b[j];
      ((float2*)dst)[c] = v;
    }
  }
}

extern "C" void kernel_launch(void* const* d_in, const int* in_sizes, int n_in,
                              void* d_out, int out_size, void* d_ws, size_t ws_size,
                              hipStream_t stream) {
  (void)in_sizes; (void)n_in; (void)out_size;
  const float* features = (const float*)d_in[0];   // 256 x 512
  const float* textf    = (const float*)d_in[1];   // 1000 x 512
  /* d_in[2] = targets: unused by the reference */
  const float* simw     = (const float*)d_in[3];   // 8192 x 1000 x 2
  const int*   pairs    = (const int*)d_in[4];     // 2000 x 2
  float* out = (float*)d_out;

  // Dead-in-reference matmul, kept faithful (WMMA) when workspace allows.
  if (ws_size >= (size_t)256 * NCLS * sizeof(float)) {
    sim_wmma_kernel<<<252, 128, 0, stream>>>(features, textf, (float*)d_ws);
  }

  sinkhorn_kernel<<<NROWS, 256, 0, stream>>>(simw, pairs, out);
}
